// MotionDiscrim_40054865002852
// MI455X (gfx1250) — matrix operand order repaired and checked by reading, not compiled, not run on hardware
//
#include <hip/hip_runtime.h>
#include <hip/hip_bf16.h>
#include <math.h>

typedef __attribute__((ext_vector_type(16))) __bf16 v16bf;
typedef __attribute__((ext_vector_type(8)))  __bf16 v8bf;
typedef __attribute__((ext_vector_type(8)))  float  v8f;

#define TM    32      // motion frames (T-1)
#define BB    256
#define CC    512
#define HW    16
#define TXT   256
#define CP    768     // C + TXT
#define EPSBN 1e-5f
#define SLOPE 0.2f

static __device__ __forceinline__ float leaky(float x){ return x > 0.f ? x : SLOPE * x; }

static __device__ __forceinline__ v8f wmma_bf16(v16bf a, v16bf b, v8f c){
  return __builtin_amdgcn_wmma_f32_16x16x32_bf16(false, a, false, b, (short)0, c, false, false);
}

// ---------------- Tensor Data Mover: 1-D contiguous global -> LDS ----------------
#if defined(__HIP_DEVICE_COMPILE__) && __has_builtin(__builtin_amdgcn_tensor_load_to_lds)
#define HAVE_TDM 1
#else
#define HAVE_TDM 0
#endif

#if HAVE_TDM
typedef __attribute__((ext_vector_type(4))) unsigned int u32x4;
typedef __attribute__((ext_vector_type(8))) int          i32x8;
typedef __attribute__((ext_vector_type(4))) int          i32x4;

// Copy n8 * 8 bytes from gsrc (8B-aligned) into LDS at lds_byte_addr.
// D# built per CDNA5 ISA (group0/group1; 2D-capable groups 2/3 zero).
static __device__ __forceinline__ void tdm_load_1d(unsigned int lds_byte_addr,
                                                   const void* gsrc,
                                                   unsigned int n8){
  unsigned long long ga = (unsigned long long)(uintptr_t)gsrc;
  u32x4 g0;
  g0[0] = 1u;                                              // count=1, user descriptor
  g0[1] = lds_byte_addr;                                   // lds_addr [63:32]
  g0[2] = (unsigned int)ga;                                // global_addr [95:64]
  g0[3] = (unsigned int)((ga >> 32) & 0x1FFFFFFull)        // global_addr [120:96]
        | (2u << 30);                                      // type=2 ("image")
  i32x8 g1;
  g1[0] = (int)(3u << 16);                                 // wg_mask=0, data_size=3 (8B)
  g1[1] = (int)((n8 & 0xFFFFu) << 16);                     // tensor_dim0[15:0] @ bits 63:48
  g1[2] = (int)(((n8 >> 16) & 0xFFFFu) | (1u << 16));      // tensor_dim0[31:16], tensor_dim1=1
  g1[3] = (int)((n8 & 0xFFFFu) << 16);                     // tile_dim0 @ bits 127:112
  g1[4] = 0;                                               // tile_dim1=0, tile_dim2=0 (1-D)
  g1[5] = (int)n8;                                         // tensor_dim0_stride[31:0]
  g1[6] = 0;
  g1[7] = 0;
  i32x4 z4 = {};
#if defined(__clang_major__) && (__clang_major__ >= 23)
  i32x8 z8 = {};
  __builtin_amdgcn_tensor_load_to_lds(g0, g1, z4, z4, z8, 0);
#else
  __builtin_amdgcn_tensor_load_to_lds(g0, g1, z4, z4, 0);
#endif
}
static __device__ __forceinline__ unsigned int lds_off(const void* p){
  // flat address in the LDS aperture: low 32 bits are the wave's LDS byte offset
  return (unsigned int)(uintptr_t)p;
}
#endif

// ---------- WMMA fragment helpers (wave32, 16x16x32 bf16, contiguous k = 16*h + s per lane half) ----------
// A fragment: element (m,k) of 16x32 tile at abase[m*ld + k], bf16 memory.
static __device__ __forceinline__ v16bf load_a_bf16(const __bf16* abase, int ld){
  int lane = threadIdx.x & 31;
  int h = lane >> 4, m = lane & 15;
  const __bf16* r = abase + (size_t)m * ld + h * 16;
  union { v16bf v; v8bf q[2]; } u;
  u.q[0] = *(const v8bf*)(r);
  u.q[1] = *(const v8bf*)(r + 8);
  return u.v;
}
// B fragment: element (k,n) of 32x16 tile stored K-contiguous per n: bbase[n*ld + k].
static __device__ __forceinline__ v16bf load_b_kcontig(const __bf16* bbase, int ld){
  int lane = threadIdx.x & 31;
  int h = lane >> 4, n = lane & 15;
  const __bf16* r = bbase + (size_t)n * ld + h * 16;
  union { v16bf v; v8bf q[2]; } u;
  u.q[0] = *(const v8bf*)(r);
  u.q[1] = *(const v8bf*)(r + 8);
  return u.v;
}
// f32-source variants for the small linear GEMM.
static __device__ __forceinline__ v16bf load_a_f32(const float* abase, int ld){
  int lane = threadIdx.x & 31;
  int h = lane >> 4, m = lane & 15;
  const float* r = abase + (size_t)m * ld;
  v16bf a;
#pragma unroll
  for (int e = 0; e < 8; ++e) a[e] = (__bf16)r[h * 16 + e];
#pragma unroll
  for (int e = 0; e < 8; ++e) a[e + 8] = (__bf16)r[h * 16 + 8 + e];
  return a;
}
static __device__ __forceinline__ v16bf load_b_kcontig_f32(const float* bbase, int ld){
  int lane = threadIdx.x & 31;
  int h = lane >> 4, n = lane & 15;
  const float* r = bbase + (size_t)n * ld + h * 16;
  v16bf b;
#pragma unroll
  for (int e = 0; e < 16; ++e) b[e] = (__bf16)r[e];
  return b;
}

// ---------------- K0: weights -> bf16, zero BN stat accumulators ----------------
__global__ void k_init(const float* Wmm, const float* Wp1,
                       __bf16* wmm_bf, __bf16* wp1_bf,
                       float* msum_zero, float* psum_zero){
  int i = blockIdx.x * 256 + threadIdx.x;
  if (i < CC * CC)      wmm_bf[i] = (__bf16)Wmm[i];
  if (i < CC * CP)      wp1_bf[i] = (__bf16)Wp1[i];
  if (i < 2 * TM * CC){ msum_zero[i] = 0.f; psum_zero[i] = 0.f; }
}

// ---------------- K1: s_lin = sent @ W_lin^T + b_lin (WMMA) ----------------
__global__ void __launch_bounds__(256) k_lin(const float* sent, const float* Wlin,
                                             const float* blin, float* s_lin){
  int wave = (blockIdx.x * 256 + threadIdx.x) >> 5;   // 0..255 tiles
  int m0 = (wave >> 4) * 16, n0 = (wave & 15) * 16;
  v8f acc = {};
  for (int k0 = 0; k0 < TXT; k0 += 32){
    v16bf a = load_a_f32(sent + (size_t)m0 * TXT + k0, TXT);
    v16bf b = load_b_kcontig_f32(Wlin + (size_t)n0 * TXT + k0, TXT);
    acc = wmma_bf16(a, b, acc);
  }
  int lane = threadIdx.x & 31;
  int h = lane >> 4, n = lane & 15;
  float bias = blin[n0 + n];
#pragma unroll
  for (int v = 0; v < 8; ++v)
    s_lin[(size_t)(m0 + h * 8 + v) * TXT + n0 + n] = acc[v] + bias;
}

// ---------------- K2: BN over batch (axis 0) + leaky -> s_bf16 ----------------
__global__ void __launch_bounds__(256) k_sstats(const float* s_lin, const float* g,
                                                const float* be, __bf16* s_bf){
  __shared__ float r1[256], r2[256];
  int j = blockIdx.x;
  int b = threadIdx.x;
  float x = s_lin[(size_t)b * TXT + j];
  r1[b] = x; r2[b] = x * x;
  __syncthreads();
  for (int s = 128; s > 0; s >>= 1){
    if (b < s){ r1[b] += r1[b + s]; r2[b] += r2[b + s]; }
    __syncthreads();
  }
  float mean = r1[0] * (1.f / BB);
  float var  = r2[0] * (1.f / BB) - mean * mean;
  float sc = g[j] * rsqrtf(var + EPSBN);
  float sh = be[j] - mean * sc;
  s_bf[(size_t)b * TXT + j] = (__bf16)leaky(x * sc + sh);
}

// ------- K3: m = (frames[t+1]-frames[t]) x W_mm^T per (t,b); WMMA + BN stats -------
__global__ void __launch_bounds__(256) k_gemm1(const float* frames, const __bf16* wmm_bf,
                                               __bf16* m_bf, float* msum, float* msumsq){
  __shared__ float  Fraw[2 * CC * HW];        // raw f32 slices: [frame t][frame t+1]
  __shared__ __bf16 Xt[HW * CC];              // transposed bf16 diff: [hw][c]
  int t = blockIdx.x >> 8, b = blockIdx.x & 255;
  const float* f0 = frames + (size_t)(t * BB + b) * CC * HW;
  const float* f1 = f0 + (size_t)BB * CC * HW;
#if HAVE_TDM
  if ((threadIdx.x >> 5) == 0){               // wave 0 drives the TDM (EXEC ignored)
    tdm_load_1d(lds_off(Fraw),                        f0, (CC * HW * 4) / 8);
    tdm_load_1d(lds_off(Fraw) + CC * HW * 4,          f1, (CC * HW * 4) / 8);
    __builtin_amdgcn_s_wait_tensorcnt(0);
  }
  __syncthreads();
  for (int e = threadIdx.x; e < CC * HW; e += 256){
    int c = e >> 4, hw = e & 15;
    Xt[hw * CC + c] = (__bf16)(Fraw[CC * HW + e] - Fraw[e]);
  }
#else
  for (int e = threadIdx.x; e < CC * HW; e += 256){
    int c = e >> 4, hw = e & 15;
    Xt[hw * CC + c] = (__bf16)(f1[e] - f0[e]);
  }
#endif
  __syncthreads();
  int wave = threadIdx.x >> 5;                // 8 waves, 4 o-tiles each
  v8f acc[4] = {};
  for (int k0 = 0; k0 < CC; k0 += 32){
    v16bf bfr = load_b_kcontig(Xt + k0, CC);
#pragma unroll
    for (int i = 0; i < 4; ++i){
      int o0 = (wave + 8 * i) * 16;
      v16bf afr = load_a_bf16(wmm_bf + (size_t)o0 * CC + k0, CC);
      acc[i] = wmma_bf16(afr, bfr, acc[i]);
    }
    if (k0 + 32 < CC)
      __builtin_prefetch(wmm_bf + (size_t)(wave * 16) * CC + k0 + 32, 0, 1);
  }
  int lane = threadIdx.x & 31;
  int h = lane >> 4, n = lane & 15;
  size_t obase = (size_t)(t * BB + b) * CC;
#pragma unroll
  for (int i = 0; i < 4; ++i){
    int o0 = (wave + 8 * i) * 16;
#pragma unroll
    for (int v = 0; v < 8; ++v){
      float val = acc[i][v];
      int o = o0 + h * 8 + v;
      m_bf[(obase + o) * HW + n] = (__bf16)val;
      float s1 = val, s2 = val * val;
#pragma unroll
      for (int d = 1; d < 16; d <<= 1){
        s1 += __shfl_xor(s1, d, 32);
        s2 += __shfl_xor(s2, d, 32);
      }
      if (n == 0){
        atomicAdd(&msum[t * CC + o],   s1);
        atomicAdd(&msumsq[t * CC + o], s2);
      }
    }
  }
}

// -------- K4/K6: finalize per-(t,c) BN scale/shift from accumulated sums --------
__global__ void k_stats(const float* sum, const float* sumsq, const float* g,
                        const float* be, float* scale, float* shift){
  int i = blockIdx.x * 256 + threadIdx.x;
  if (i >= TM * CC) return;
  int c = i & (CC - 1);
  float mean = sum[i]   * (1.f / (BB * HW));
  float var  = sumsq[i] * (1.f / (BB * HW)) - mean * mean;
  float sc = g[c] * rsqrtf(var + EPSBN);
  scale[i] = sc;
  shift[i] = be[c] - mean * sc;
}

// --- K5: p = [leaky(bn(m)) ; s] x W_p1^T per (t,b); WMMA K=768 + BN stats ---
__global__ void __launch_bounds__(256) k_gemm2(const __bf16* m_bf, const __bf16* s_bf,
                                               const __bf16* wp1_bf, const float* mscale,
                                               const float* mshift, __bf16* p_bf,
                                               float* psum, float* psumsq){
  __shared__ __bf16 Mraw[CC * HW];            // raw m slice (TDM target)
  __shared__ __bf16 Mst[HW * CP];             // [hw][c], c in [0,768)
  int t = blockIdx.x >> 8, b = blockIdx.x & 255;
  const __bf16* msrc = m_bf + (size_t)(t * BB + b) * CC * HW;
#if HAVE_TDM
  if ((threadIdx.x >> 5) == 0){
    tdm_load_1d(lds_off(Mraw), msrc, (CC * HW * 2) / 8);
    __builtin_amdgcn_s_wait_tensorcnt(0);
  }
  __syncthreads();
  for (int e = threadIdx.x; e < CC * HW; e += 256){
    int c = e >> 4, hw = e & 15;
    float v = (float)Mraw[e] * mscale[t * CC + c] + mshift[t * CC + c];
    Mst[hw * CP + c] = (__bf16)leaky(v);
  }
#else
  for (int e = threadIdx.x; e < CC * HW; e += 256){
    int c = e >> 4, hw = e & 15;
    float v = (float)msrc[e] * mscale[t * CC + c] + mshift[t * CC + c];
    Mst[hw * CP + c] = (__bf16)leaky(v);
  }
#endif
  for (int e = threadIdx.x; e < TXT * HW; e += 256){
    int j = e >> 4, hw = e & 15;                // broadcast s over hw
    Mst[hw * CP + CC + j] = s_bf[(size_t)b * TXT + j];
  }
  __syncthreads();
  int wave = threadIdx.x >> 5;
  v8f acc[4] = {};
  for (int k0 = 0; k0 < CP; k0 += 32){
    v16bf bfr = load_b_kcontig(Mst + k0, CP);
#pragma unroll
    for (int i = 0; i < 4; ++i){
      int o0 = (wave + 8 * i) * 16;
      v16bf afr = load_a_bf16(wp1_bf + (size_t)o0 * CP + k0, CP);
      acc[i] = wmma_bf16(afr, bfr, acc[i]);
    }
    if (k0 + 32 < CP)
      __builtin_prefetch(wp1_bf + (size_t)(wave * 16) * CP + k0 + 32, 0, 1);
  }
  int lane = threadIdx.x & 31;
  int h = lane >> 4, n = lane & 15;
  size_t obase = (size_t)(t * BB + b) * CC;
#pragma unroll
  for (int i = 0; i < 4; ++i){
    int o0 = (wave + 8 * i) * 16;
#pragma unroll
    for (int v = 0; v < 8; ++v){
      float val = acc[i][v];
      int o = o0 + h * 8 + v;
      p_bf[(obase + o) * HW + n] = (__bf16)val;
      float s1 = val, s2 = val * val;
#pragma unroll
      for (int d = 1; d < 16; d <<= 1){
        s1 += __shfl_xor(s1, d, 32);
        s2 += __shfl_xor(s2, d, 32);
      }
      if (n == 0){
        atomicAdd(&psum[t * CC + o],   s1);
        atomicAdd(&psumsq[t * CC + o], s2);
      }
    }
  }
}

// ---------- K7: out[t,b] = sigmoid( sum leaky(bn(p)) * W_p2 ) ----------
__global__ void __launch_bounds__(256) k_final(const __bf16* p_bf, const float* pscale,
                                               const float* pshift, const float* Wp2,
                                               float* out){
  __shared__ float red[256];
  int t = blockIdx.x >> 8, b = blockIdx.x & 255;
  const __bf16* src = p_bf + (size_t)(t * BB + b) * CC * HW;
  float acc = 0.f;
  for (int e = threadIdx.x; e < CC * HW; e += 256){
    int c = e >> 4;
    float v = (float)src[e] * pscale[t * CC + c] + pshift[t * CC + c];
    acc += leaky(v) * Wp2[e];
  }
  red[threadIdx.x] = acc;
  __syncthreads();
  for (int s = 128; s > 0; s >>= 1){
    if (threadIdx.x < s) red[threadIdx.x] += red[threadIdx.x + s];
    __syncthreads();
  }
  if (threadIdx.x == 0)
    out[t * BB + b] = 1.f / (1.f + __expf(-red[0]));
}

extern "C" void kernel_launch(void* const* d_in, const int* in_sizes, int n_in,
                              void* d_out, int out_size, void* d_ws, size_t ws_size,
                              hipStream_t stream){
  (void)in_sizes; (void)n_in; (void)out_size; (void)ws_size;
  const float* frames = (const float*)d_in[0];
  const float* sent   = (const float*)d_in[1];
  const float* Wlin   = (const float*)d_in[2];
  const float* blin   = (const float*)d_in[3];
  const float* g_s    = (const float*)d_in[4];
  const float* b_s    = (const float*)d_in[5];
  const float* Wmm    = (const float*)d_in[6];
  const float* g_m    = (const float*)d_in[7];
  const float* b_m    = (const float*)d_in[8];
  const float* Wp1    = (const float*)d_in[9];
  const float* g_p    = (const float*)d_in[10];
  const float* b_p    = (const float*)d_in[11];
  const float* Wp2    = (const float*)d_in[12];
  float* out = (float*)d_out;

  char* ws = (char*)d_ws;
  size_t off = 0;
  auto alloc = [&](size_t bytes) -> void* {
    void* p = ws + off;
    off = (off + bytes + 255) & ~(size_t)255;
    return p;
  };
  float*  s_lin  = (float*) alloc((size_t)BB * TXT * 4);
  __bf16* s_bf   = (__bf16*)alloc((size_t)BB * TXT * 2);
  __bf16* wmm_bf = (__bf16*)alloc((size_t)CC * CC * 2);
  __bf16* wp1_bf = (__bf16*)alloc((size_t)CC * CP * 2);
  float*  msum   = (float*) alloc((size_t)TM * CC * 4);   // msum..msumsq contiguous
  float*  msumsq = (float*) alloc((size_t)TM * CC * 4);
  float*  mscale = (float*) alloc((size_t)TM * CC * 4);
  float*  mshift = (float*) alloc((size_t)TM * CC * 4);
  float*  psum   = (float*) alloc((size_t)TM * CC * 4);   // psum..psumsq contiguous
  float*  psumsq = (float*) alloc((size_t)TM * CC * 4);
  float*  pscale = (float*) alloc((size_t)TM * CC * 4);
  float*  pshift = (float*) alloc((size_t)TM * CC * 4);
  __bf16* m_bf   = (__bf16*)alloc((size_t)TM * BB * CC * HW * 2);
  __bf16* p_bf   = (__bf16*)alloc((size_t)TM * BB * CC * HW * 2);

  k_init  <<<(CC * CP + 255) / 256, 256, 0, stream>>>(Wmm, Wp1, wmm_bf, wp1_bf, msum, psum);
  k_lin   <<<32, 256, 0, stream>>>(sent, Wlin, blin, s_lin);
  k_sstats<<<TXT, 256, 0, stream>>>(s_lin, g_s, b_s, s_bf);
  k_gemm1 <<<TM * BB, 256, 0, stream>>>(frames, wmm_bf, m_bf, msum, msumsq);
  k_stats <<<(TM * CC + 255) / 256, 256, 0, stream>>>(msum, msumsq, g_m, b_m, mscale, mshift);
  k_gemm2 <<<TM * BB, 256, 0, stream>>>(m_bf, s_bf, wp1_bf, mscale, mshift, p_bf, psum, psumsq);
  k_stats <<<(TM * CC + 255) / 256, 256, 0, stream>>>(psum, psumsq, g_p, b_p, pscale, pshift);
  k_final <<<TM * BB, 256, 0, stream>>>(p_bf, pscale, pshift, Wp2, out);
}
